// SFSG_24721831756527
// MI455X (gfx1250) — matrix operand order repaired and checked by reading, not compiled, not run on hardware
//
#include <hip/hip_runtime.h>
#include <hip/hip_bf16.h>
#include <math.h>

#define NU 20000
#define NI 10000
#define DD 64
#define NBITS 4
#define KNNK 10
#define NNZ 500000
#define NNODES (NU + NI)

typedef __attribute__((ext_vector_type(16))) _Float16 v16h;
typedef __attribute__((ext_vector_type(8)))  float    v8f;

// ---------------------------------------------------------------------------
// Generic WMMA GEMM: C[M,N] = A[M,K] @ B[K,N] (+bias) (optional relu)
// A,B f32 row-major -> converted to f16 on load; f32 accumulate.
// One wave (32 threads) per 16x32 output tile (A fragment shared by two
// B fragments -> halves A traffic, 2 WMMAs per K-chunk).
// Requires M%16==0, N%32==0, K%32==0 (true for every GEMM in this pipeline).
// Fragment layouts per CDNA5 ISA 7.12.2.
// ---------------------------------------------------------------------------
__global__ void __launch_bounds__(32)
k_gemm_wmma(const float* __restrict__ A, const float* __restrict__ B,
            const float* __restrict__ bias, float* __restrict__ C,
            int M, int N, int K, int relu)
{
    const int n0 = blockIdx.x * 32;
    const int m0 = blockIdx.y * 16;
    const int lane = threadIdx.x & 31;
    const int lh = lane >> 4;     // lane-half select
    const int ll = lane & 15;
    v8f acc0 = {}, acc1 = {};
    const float* Arow = A + (size_t)(m0 + ll) * K;
    for (int k0 = 0; k0 < K; k0 += 32) {
        if (k0 + 64 <= K) __builtin_prefetch(Arow + k0 + 64, 0, 1);
        v16h a, b0, b1;
#pragma unroll
        for (int j = 0; j < 16; ++j) {
            // A 16x32 f16 layout: lane<16 -> K in {0..7,16..23}; lane>=16 -> +8
            int ka = k0 + (j & 7) + ((j >> 3) << 4) + (lh << 3);
            a[j] = (_Float16)Arow[ka];
            // B 32x16 f16 layout: same K order as A (lane<16 -> K in {0..7,16..23}; lane>=16 -> +8)
            int kb = k0 + (j & 7) + ((j >> 3) << 4) + (lh << 3);
            const float* Brow = B + (size_t)kb * N + n0;
            b0[j] = (_Float16)Brow[ll];
            b1[j] = (_Float16)Brow[16 + ll];
        }
        acc0 = __builtin_amdgcn_wmma_f32_16x16x32_f16(false, a, false, b0,
                                                      (short)0, acc0, false, false);
        acc1 = __builtin_amdgcn_wmma_f32_16x16x32_f16(false, a, false, b1,
                                                      (short)0, acc1, false, false);
    }
    const int col0 = n0 + ll;
    const int col1 = n0 + 16 + ll;
    const float bv0 = bias ? bias[col0] : 0.0f;
    const float bv1 = bias ? bias[col1] : 0.0f;
#pragma unroll
    for (int r = 0; r < 8; ++r) {
        int row = m0 + r + lh * 8;   // C layout: vgpr r -> M = r (+8 for lanes 16..31)
        float v0 = acc0[r] + bv0;
        float v1 = acc1[r] + bv1;
        if (relu) { v0 = fmaxf(v0, 0.0f); v1 = fmaxf(v1, 0.0f); }
        C[(size_t)row * N + col0] = v0;
        C[(size_t)row * N + col1] = v1;
    }
}

// ---------------------------------------------------------------------------
// UI bipartite-graph LightGCN kernels
// ---------------------------------------------------------------------------
__global__ void k_deg(const int* __restrict__ uu, const int* __restrict__ ii,
                      float* __restrict__ deg)
{
    int e = blockIdx.x * blockDim.x + threadIdx.x;
    if (e >= NNZ) return;
    atomicAdd(&deg[uu[e]], 1.0f);
    atomicAdd(&deg[NU + ii[e]], 1.0f);
}

__global__ void k_dinv(float* __restrict__ deg)
{
    int i = blockIdx.x * blockDim.x + threadIdx.x;
    if (i >= NNODES) return;
    deg[i] = powf(deg[i] + 1e-7f, -0.5f);
}

__global__ void k_spmm_ui(const int* __restrict__ uu, const int* __restrict__ ii,
                          const float* __restrict__ dinv,
                          const float* __restrict__ hu, const float* __restrict__ hi,
                          float* __restrict__ out)
{
    long long t = (long long)blockIdx.x * blockDim.x + threadIdx.x;
    if (t >= (long long)2 * NNZ * 16) return;
    int slot = (int)(t >> 4);
    int d4 = ((int)t & 15) * 4;
    int e = (slot >= NNZ) ? slot - NNZ : slot;
    int r, c;
    if (slot < NNZ) { r = uu[e]; c = NU + ii[e]; }
    else            { r = NU + ii[e]; c = uu[e]; }
    float val = dinv[r] * dinv[c];
    const float* src = (c < NU) ? (hu + (size_t)c * DD) : (hi + (size_t)(c - NU) * DD);
#pragma unroll
    for (int q = 0; q < 4; ++q)
        atomicAdd(&out[(size_t)r * DD + d4 + q], val * src[d4 + q]);
}

__global__ void k_combine_ui(const float* __restrict__ ue, const float* __restrict__ ie,
                             const float* __restrict__ h1, const float* __restrict__ h2,
                             float* __restrict__ acc)
{
    long long idx = (long long)blockIdx.x * blockDim.x + threadIdx.x;
    if (idx >= (long long)NNODES * DD) return;
    long long node = idx / DD;
    float ego = (node < NU) ? ue[idx] : ie[idx - (long long)NU * DD];
    acc[idx] = (1.0f / 3.0f) * ego + (2.0f / 3.0f) * h1[idx] + h2[idx];
}

// ---------------------------------------------------------------------------
// Encoder helper kernels
// ---------------------------------------------------------------------------
// x[s,n,d] = x4[n, s*64+d] + pos_encoding(s,d)
__global__ void k_build_x(const float* __restrict__ x4, float* __restrict__ x)
{
    long long idx = (long long)blockIdx.x * blockDim.x + threadIdx.x;
    if (idx >= (long long)NBITS * NI * DD) return;
    int d = (int)(idx % DD);
    long long r = idx / DD;
    int n = (int)(r % NI);
    int s = (int)(r / NI);
    int i2 = d & ~1;
    float freq = expf(-(float)i2 * (logf(10000.0f) / (float)DD));
    float pe = (d & 1) ? cosf((float)s * freq) : sinf((float)s * freq);
    x[idx] = x4[(size_t)n * (NBITS * DD) + s * DD + d] + pe;
}

// Per-(item,head) 4-token attention: qkv (s,n,192) -> o (s,n,64)
__global__ void k_attn(const float* __restrict__ qkv, float* __restrict__ o)
{
    int t = blockIdx.x * blockDim.x + threadIdx.x;
    if (t >= NI * 4) return;
    int n = t >> 2, h = t & 3;
    float kk[4][16], vv[4][16];
#pragma unroll
    for (int s2 = 0; s2 < 4; ++s2) {
        const float* base = qkv + ((size_t)(s2 * NI + n)) * 192 + h * 16;
#pragma unroll
        for (int d = 0; d < 16; ++d) { kk[s2][d] = base[64 + d]; vv[s2][d] = base[128 + d]; }
    }
#pragma unroll
    for (int s = 0; s < 4; ++s) {
        const float* qb = qkv + ((size_t)(s * NI + n)) * 192 + h * 16;
        float qv[16];
#pragma unroll
        for (int d = 0; d < 16; ++d) qv[d] = qb[d];
        float sc[4];
#pragma unroll
        for (int t2 = 0; t2 < 4; ++t2) {
            float dot = 0.0f;
#pragma unroll
            for (int d = 0; d < 16; ++d) dot += qv[d] * kk[t2][d];
            sc[t2] = dot * 0.25f;   // / sqrt(16)
        }
        float mx = fmaxf(fmaxf(sc[0], sc[1]), fmaxf(sc[2], sc[3]));
        float se = 0.0f;
#pragma unroll
        for (int t2 = 0; t2 < 4; ++t2) { sc[t2] = expf(sc[t2] - mx); se += sc[t2]; }
        float rinv = 1.0f / se;
        float* ob = o + ((size_t)(s * NI + n)) * DD + h * 16;
#pragma unroll
        for (int d = 0; d < 16; ++d) {
            float acc = 0.0f;
#pragma unroll
            for (int t2 = 0; t2 < 4; ++t2) acc += sc[t2] * vv[t2][d];
            ob[d] = acc * rinv;
        }
    }
}

// In-place residual LayerNorm over 64 dims: x = LN(x + o) * g + b. One wave/row.
__global__ void __launch_bounds__(32)
k_ln(float* __restrict__ x, const float* __restrict__ o,
     const float* __restrict__ g, const float* __restrict__ b, int rows)
{
    int row = blockIdx.x;
    if (row >= rows) return;
    int lane = threadIdx.x;
    size_t base = (size_t)row * DD;
    float v0 = x[base + lane] + o[base + lane];
    float v1 = x[base + 32 + lane] + o[base + 32 + lane];
    float s = v0 + v1, ss = v0 * v0 + v1 * v1;
    for (int m = 16; m >= 1; m >>= 1) {
        s  += __shfl_xor(s,  m, 32);
        ss += __shfl_xor(ss, m, 32);
    }
    float mean = s * (1.0f / 64.0f);
    float var  = ss * (1.0f / 64.0f) - mean * mean;
    float inv  = rsqrtf(var + 1e-5f);
    x[base + lane]      = (v0 - mean) * inv * g[lane]      + b[lane];
    x[base + 32 + lane] = (v1 - mean) * inv * g[32 + lane] + b[32 + lane];
}

// full[n, s*64+d] = x[s,n,d]
__global__ void k_build_full(const float* __restrict__ x, float* __restrict__ full)
{
    long long idx = (long long)blockIdx.x * blockDim.x + threadIdx.x;
    if (idx >= (long long)NBITS * NI * DD) return;
    int d = (int)(idx % DD);
    long long r = idx / DD;
    int n = (int)(r % NI);
    int s = (int)(r / NI);
    full[(size_t)n * (NBITS * DD) + s * DD + d] = x[idx];
}

// L2-normalize each 256-dim row and emit f16 for the WMMA sim GEMM. One wave/row.
__global__ void __launch_bounds__(32)
k_rownorm_f16(const float* __restrict__ full, _Float16* __restrict__ fn)
{
    int row = blockIdx.x;
    if (row >= NI) return;
    int lane = threadIdx.x;
    const float* src = full + (size_t)row * 256;
    float vals[8]; float ss = 0.0f;
#pragma unroll
    for (int j = 0; j < 8; ++j) { float v = src[j * 32 + lane]; vals[j] = v; ss += v * v; }
    for (int m = 16; m >= 1; m >>= 1) ss += __shfl_xor(ss, m, 32);
    float inv = rsqrtf(ss);
    _Float16* dst = fn + (size_t)row * 256;
#pragma unroll
    for (int j = 0; j < 8; ++j) dst[j * 32 + lane] = (_Float16)(vals[j] * inv);
}

// Fused cosine-sim GEMM + top-10. One wave owns 32 rows: two A-fragment sets
// (32x256 f16) stay resident in VGPRs, every B tile loaded once serves both
// row tiles (16 WMMAs per column tile), and each of the 32 lanes maintains the
// top-10 for exactly one row. sim never touches global memory (saves 2x800MB
// of HBM traffic vs materializing the 10000x10000 matrix).
__global__ void __launch_bounds__(32)
k_knn_topk(const _Float16* __restrict__ F, int* __restrict__ ind)
{
    __shared__ float tile[32][17];
    const int m0 = blockIdx.x * 32;
    const int lane = threadIdx.x & 31;
    const int lh = lane >> 4;
    const int ll = lane & 15;

    // rows for the two resident A tiles (clamped: last block may be partial;
    // EXEC must stay all-ones for WMMA, so we compute garbage rows but never
    // write their results)
    int rowA0 = m0 + ll;            if (rowA0 >= NI) rowA0 = NI - 1;
    int rowA1 = m0 + 16 + ll;       if (rowA1 >= NI) rowA1 = NI - 1;

    v16h afr0[8], afr1[8];
#pragma unroll
    for (int c = 0; c < 8; ++c) {
#pragma unroll
        for (int j = 0; j < 16; ++j) {
            int ka = c * 32 + (j & 7) + ((j >> 3) << 4) + (lh << 3);
            afr0[c][j] = F[(size_t)rowA0 * 256 + ka];
            afr1[c][j] = F[(size_t)rowA1 * 256 + ka];
        }
    }

    float best[KNNK]; int bidx[KNNK];
#pragma unroll
    for (int p = 0; p < KNNK; ++p) { best[p] = -1e30f; bidx[p] = 0; }

    for (int n0 = 0; n0 < NI; n0 += 16) {
        v8f acc0 = {}, acc1 = {};
#pragma unroll
        for (int c = 0; c < 8; ++c) {
            v16h b;
#pragma unroll
            for (int j = 0; j < 16; ++j) {
                int kb = c * 32 + (j & 7) + ((j >> 3) << 4) + (lh << 3);
                b[j] = F[(size_t)(n0 + ll) * 256 + kb];   // B = F^T tile
            }
            acc0 = __builtin_amdgcn_wmma_f32_16x16x32_f16(false, afr0[c], false, b,
                                                          (short)0, acc0, false, false);
            acc1 = __builtin_amdgcn_wmma_f32_16x16x32_f16(false, afr1[c], false, b,
                                                          (short)0, acc1, false, false);
        }
#pragma unroll
        for (int r = 0; r < 8; ++r) {
            tile[r + lh * 8][ll]      = acc0[r];
            tile[16 + r + lh * 8][ll] = acc1[r];
        }
        __syncthreads();
        {
#pragma unroll
            for (int j = 0; j < 16; ++j) {
                float v = tile[lane][j];
                if (v > best[KNNK - 1]) {
                    int id = n0 + j;
#pragma unroll
                    for (int p = 0; p < KNNK; ++p) {
                        if (v > best[p]) {
                            float tv = best[p]; int ti = bidx[p];
                            best[p] = v; bidx[p] = id; v = tv; id = ti;
                        }
                    }
                }
            }
        }
        __syncthreads();
    }
    if (m0 + lane < NI) {
#pragma unroll
        for (int p = 0; p < KNNK; ++p) ind[(size_t)(m0 + lane) * KNNK + p] = bidx[p];
    }
}

// contrib[i,d] = (1/(K+1e-7)) * sum_p i_g[ind[i][p], d] + proj[i,d]
__global__ void k_contrib(const int* __restrict__ ind, const float* __restrict__ ig,
                          const float* __restrict__ proj, float* __restrict__ contrib)
{
    long long idx = (long long)blockIdx.x * blockDim.x + threadIdx.x;
    if (idx >= (long long)NI * DD) return;
    int i = (int)(idx / DD), d = (int)(idx % DD);
    float s = 0.0f;
#pragma unroll
    for (int p = 0; p < KNNK; ++p)
        s += ig[(size_t)ind[(size_t)i * KNNK + p] * DD + d];
    float rinv = powf((float)KNNK + 1e-7f, -0.5f);
    contrib[idx] = rinv * rinv * s + proj[idx];
}

__global__ void k_final(const float* __restrict__ acc, const float* __restrict__ ci,
                        const float* __restrict__ ct, const float* __restrict__ mw,
                        float* __restrict__ out)
{
    long long idx = (long long)blockIdx.x * blockDim.x + threadIdx.x;
    if (idx >= (long long)NNODES * DD) return;
    long long node = idx / DD;
    if (node < NU) { out[idx] = acc[idx]; return; }
    float m0 = mw[0], m1 = mw[1];
    float mx = fmaxf(m0, m1);
    float e0 = expf(m0 - mx), e1 = expf(m1 - mx);
    float w0 = e0 / (e0 + e1), w1 = e1 / (e0 + e1);
    long long j = idx - (long long)NU * DD;
    out[idx] = acc[idx] + w0 * ci[j] + w1 * ct[j];
}

// ---------------------------------------------------------------------------
// Host side
// ---------------------------------------------------------------------------
struct ModalParams {
    const float *W_trs, *b_trs, *W_con, *b_con, *Wqkv, *bqkv, *Wo, *bo;
    const float *ln1g, *ln1b, *ln2g, *ln2b, *W1, *b1, *W2, *b2, *W_lin, *b_lin;
};

static void fillp(ModalParams& p, void* const* d_in, int b, bool sorted)
{
    const float* a[18];
    for (int i = 0; i < 18; ++i) a[i] = (const float*)d_in[b + i];
    if (!sorted) {  // dict insertion order
        p.W_trs = a[0];  p.b_trs = a[1];  p.W_con = a[2];  p.b_con = a[3];
        p.Wqkv  = a[4];  p.bqkv  = a[5];  p.Wo    = a[6];  p.bo    = a[7];
        p.ln1g  = a[8];  p.ln1b  = a[9];  p.ln2g  = a[10]; p.ln2b  = a[11];
        p.W1    = a[12]; p.b1    = a[13]; p.W2    = a[14]; p.b2    = a[15];
        p.W_lin = a[16]; p.b_lin = a[17];
    } else {        // jax pytree (sorted-key) order
        p.W1    = a[0];  p.W2    = a[1];  p.W_con = a[2];  p.W_lin = a[3];
        p.W_trs = a[4];  p.Wo    = a[5];  p.Wqkv  = a[6];
        p.b1    = a[7];  p.b2    = a[8];  p.b_con = a[9];  p.b_lin = a[10];
        p.b_trs = a[11]; p.bo    = a[12]; p.bqkv  = a[13];
        p.ln1b  = a[14]; p.ln1g  = a[15]; p.ln2b  = a[16]; p.ln2g  = a[17];
    }
}

static inline void gemm(hipStream_t st, const float* A, const float* B,
                        const float* bias, float* C, int M, int N, int K, int relu)
{
    dim3 grid(N / 32, M / 16);
    k_gemm_wmma<<<grid, 32, 0, st>>>(A, B, bias, C, M, N, K, relu);
}

// Workspace layout (float offsets)
#define OFF_DINV   0LL
#define OFF_H1     30016LL
#define OFF_H2     1950016LL
#define OFF_ACC    3870016LL
#define OFF_FEATS  5790016LL
#define OFF_X      6430016LL
#define OFF_QKV    8990016LL     // 7,680,000 floats; reused for x4 / o2 / ffout / FULL / FNORM
#define OFF_O      16670016LL
#define OFF_IND    19230016LL
#define OFF_CI     19330016LL
#define OFF_CT     19970016LL

extern "C" void kernel_launch(void* const* d_in, const int* in_sizes, int n_in,
                              void* d_out, int out_size, void* d_ws, size_t ws_size,
                              hipStream_t stream)
{
    // --- resolve input ordering (insertion vs sorted pytree flattening) ---
    bool top_sorted = (in_sizes[0] != NU * DD);
    int i_user, i_item, i_mw, i_img, i_txt, pbase_i, pbase_t, i_uu, i_ui;
    if (!top_sorted) {
        i_user = 0; i_item = 1; i_mw = 2; i_img = 3; i_txt = 4;
        pbase_i = 5; pbase_t = 23; i_uu = 41; i_ui = 42;
    } else {
        i_img = 0; pbase_i = 1; i_item = 19; i_mw = 20; i_txt = 21;
        pbase_t = 22; i_ui = 40; i_uu = 41; i_user = 42;
    }
    bool nested_sorted = (in_sizes[pbase_i] == DD * DD);  // W1 first if sorted

    const float* user_emb = (const float*)d_in[i_user];
    const float* item_emb = (const float*)d_in[i_item];
    const float* mweight  = (const float*)d_in[i_mw];
    const float* img_raw  = (const float*)d_in[i_img];
    const float* txt_raw  = (const float*)d_in[i_txt];
    const int*   uu       = (const int*)d_in[i_uu];
    const int*   ui       = (const int*)d_in[i_ui];
    ModalParams Pimg, Ptxt;
    fillp(Pimg, d_in, pbase_i, nested_sorted);
    fillp(Ptxt, d_in, pbase_t, nested_sorted);

    float* W = (float*)d_ws;
    float* DINV  = W + OFF_DINV;
    float* H1    = W + OFF_H1;
    float* H2    = W + OFF_H2;
    float* ACC   = W + OFF_ACC;
    float* FEATS = W + OFF_FEATS;
    float* XBUF  = W + OFF_X;
    float* QKV   = W + OFF_QKV;
    float* OBUF  = W + OFF_O;
    int*   IND   = (int*)(W + OFF_IND);
    float* CI    = W + OFF_CI;
    float* CT    = W + OFF_CT;
    float* FULL  = QKV;                                   // (NI,256) alias
    _Float16* FNORM = (_Float16*)(QKV + 2560000);         // (NI,256) f16 alias
    float* out = (float*)d_out;

    // ---------------- UI graph propagation ----------------
    hipMemsetAsync(DINV, 0, NNODES * sizeof(float), stream);
    hipMemsetAsync(H1, 0, (size_t)NNODES * DD * sizeof(float), stream);
    hipMemsetAsync(H2, 0, (size_t)NNODES * DD * sizeof(float), stream);
    k_deg<<<(NNZ + 255) / 256, 256, 0, stream>>>(uu, ui, DINV);
    k_dinv<<<(NNODES + 255) / 256, 256, 0, stream>>>(DINV);
    long long spmm_t = (long long)2 * NNZ * 16;
    k_spmm_ui<<<(unsigned)((spmm_t + 255) / 256), 256, 0, stream>>>(
        uu, ui, DINV, user_emb, item_emb, H1);
    k_spmm_ui<<<(unsigned)((spmm_t + 255) / 256), 256, 0, stream>>>(
        uu, ui, DINV, H1, H1 + (size_t)NU * DD, H2);
    long long nall = (long long)NNODES * DD;
    k_combine_ui<<<(unsigned)((nall + 255) / 256), 256, 0, stream>>>(
        user_emb, item_emb, H1, H2, ACC);
    const float* i_g = ACC + (size_t)NU * DD;

    // ---------------- modality encoders + kNN graphs ----------------
    long long nx = (long long)NBITS * NI * DD;
    for (int m = 0; m < 2; ++m) {
        const ModalParams& P = m ? Ptxt : Pimg;
        const float* raw = m ? txt_raw : img_raw;
        int rawdim = m ? 384 : 4096;
        float* contrib = m ? CT : CI;

        gemm(stream, raw, P.W_trs, P.b_trs, FEATS, NI, DD, rawdim, 0);          // feats
        gemm(stream, FEATS, P.W_con, P.b_con, QKV, NI, NBITS * DD, DD, 0);      // x4 -> QKV
        k_build_x<<<(unsigned)((nx + 255) / 256), 256, 0, stream>>>(QKV, XBUF); // x = x4 + pe
        gemm(stream, XBUF, P.Wqkv, P.bqkv, QKV, NBITS * NI, 3 * DD, DD, 0);     // qkv
        k_attn<<<(NI * 4 + 255) / 256, 256, 0, stream>>>(QKV, OBUF);            // attention
        gemm(stream, OBUF, P.Wo, P.bo, QKV, NBITS * NI, DD, DD, 0);             // o2 -> QKV
        k_ln<<<NBITS * NI, 32, 0, stream>>>(XBUF, QKV, P.ln1g, P.ln1b, NBITS * NI);
        gemm(stream, XBUF, P.W1, P.b1, OBUF, NBITS * NI, DD, DD, 1);            // relu ff1
        gemm(stream, OBUF, P.W2, P.b2, QKV, NBITS * NI, DD, DD, 0);             // ff2 -> QKV
        k_ln<<<NBITS * NI, 32, 0, stream>>>(XBUF, QKV, P.ln2g, P.ln2b, NBITS * NI);
        k_build_full<<<(unsigned)((nx + 255) / 256), 256, 0, stream>>>(XBUF, FULL);
        k_rownorm_f16<<<NI, 32, 0, stream>>>(FULL, FNORM);
        k_knn_topk<<<(NI + 31) / 32, 32, 0, stream>>>(FNORM, IND);              // fused sim+topk
        gemm(stream, FULL, P.W_lin, P.b_lin, FEATS, NI, DD, NBITS * DD, 0);     // proj
        long long ni = (long long)NI * DD;
        k_contrib<<<(unsigned)((ni + 255) / 256), 256, 0, stream>>>(IND, i_g, FEATS, contrib);
    }

    // ---------------- final combine ----------------
    k_final<<<(unsigned)((nall + 255) / 256), 256, 0, stream>>>(ACC, CI, CT, mweight, out);
}
